// StructureAware_LoRADecoder_MoE_31550829756914
// MI455X (gfx1250) — matrix operand order repaired and checked by reading, not compiled, not run on hardware
//
#include <hip/hip_runtime.h>
#include <cstdint>
#include <cstddef>

// ---------------------------------------------------------------- constants
constexpr int B_  = 32;
constexpr int N_  = 288;
constexpr int NTOK = B_ * N_;        // 9216 tokens
constexpr int DIN  = 512;
constexpr int HIDD = 128;
constexpr int OUTD = 16384;
constexpr int NE   = 8;
constexpr size_t A_ELEMS = 75497472ull;           // 32*3*24*2*16*1024
constexpr size_t AUX_IDX = 2ull * A_ELEMS;        // scalar aux slot in d_out

// workspace layout (bytes)
constexpr size_t WS_COUNTS = 0;          // 16 ints
constexpr size_t WS_SEL    = 64;         // 9216*2 ints
constexpr size_t WS_GATES  = 73792;      // 9216*2 f32
constexpr size_t WS_PROBS  = 147520;     // 9216*8 f32
constexpr size_t WS_LISTS  = 442432;     // 16*9216 ints
constexpr size_t WS_HW     = 1032256;    // 16*9216*128 bf16 (~37.7 MB)
constexpr size_t WS_W1BF   = 38780992;   // 8*512*128 bf16   (1 MB)
constexpr size_t WS_W2BF   = 39829568;   // 8*128*16384 bf16 (33.5 MB)
constexpr int    W1_ELEMS  = NE * DIN * HIDD;     // 524288
constexpr int    W2_ELEMS  = NE * HIDD * OUTD;    // 16777216

// ---------------------------------------------------------------- types
typedef __attribute__((ext_vector_type(16))) __bf16 v16bf;
typedef __attribute__((ext_vector_type(8)))  float  v8f;

union FragAB { v16bf v; unsigned int u[8]; };

__device__ __forceinline__ unsigned short f2bf(float f) {
    unsigned int u = __float_as_uint(f);
    unsigned int r = (u + 0x7FFFu + ((u >> 16) & 1u)) >> 16;   // RNE
    return (unsigned short)r;
}
__device__ __forceinline__ unsigned int pack2(float lo, float hi) {
    return (unsigned int)f2bf(lo) | ((unsigned int)f2bf(hi) << 16);
}

// ---------------------------------------------------------------- 0) zero counts
__global__ void zero_counts_kernel(int* counts) {
    if (threadIdx.x < 16) counts[threadIdx.x] = 0;
}

// ---------------------------------------------------------------- 0b) weight f32 -> bf16 pre-conversion (cold, once per launch)
__global__ __launch_bounds__(256)
void cvt_w_kernel(const float* __restrict__ src, unsigned short* __restrict__ dst,
                  int nElem) {
    int i = (blockIdx.x * blockDim.x + threadIdx.x) * 8;
    if (i >= nElem) return;
    float4 f0 = *(const float4*)(src + i);
    float4 f1 = *(const float4*)(src + i + 4);
    uint4 q;
    q.x = pack2(f0.x, f0.y); q.y = pack2(f0.z, f0.w);
    q.z = pack2(f1.x, f1.y); q.w = pack2(f1.z, f1.w);
    *(uint4*)(dst + i) = q;
}

// ---------------------------------------------------------------- 1) router
__global__ void router_kernel(const float* __restrict__ z,
                              const float* __restrict__ emb,
                              const float* __restrict__ Wr,
                              const float* __restrict__ br,
                              int* counts, int* sel, float* gates,
                              float* probs, int* lists) {
    int t = blockIdx.x * blockDim.x + threadIdx.x;
    if (t >= NTOK) return;
    int b = t / N_, n = t % N_;
    const float* zr = z + ((size_t)b * N_ + n) * 256;
    const float* er = emb + (size_t)n * 256;

    float lg[NE];
    #pragma unroll
    for (int e = 0; e < NE; ++e) lg[e] = br[e];
    for (int c = 0; c < 256; ++c) {
        float xc = zr[c];
        const float* w = Wr + c * NE;
        #pragma unroll
        for (int e = 0; e < NE; ++e) lg[e] += xc * w[e];
    }
    for (int c = 0; c < 256; ++c) {
        float xc = er[c];
        const float* w = Wr + (256 + c) * NE;
        #pragma unroll
        for (int e = 0; e < NE; ++e) lg[e] += xc * w[e];
    }
    #pragma unroll
    for (int e = 0; e < NE; ++e) lg[e] *= (1.0f / 1.5f);   // / TAU

    // top-2 (ties -> lowest index, like jax top_k)
    float v0 = -3.4e38f, v1 = -3.4e38f; int i0 = 0, i1 = 0;
    #pragma unroll
    for (int e = 0; e < NE; ++e) {
        float v = lg[e];
        if (v > v0) { v1 = v0; i1 = i0; v0 = v; i0 = e; }
        else if (v > v1) { v1 = v; i1 = e; }
    }
    // full softmax for aux
    float se = 0.f, ex[NE];
    #pragma unroll
    for (int e = 0; e < NE; ++e) { ex[e] = __expf(lg[e] - v0); se += ex[e]; }
    float inv = 1.f / se;
    #pragma unroll
    for (int e = 0; e < NE; ++e) probs[(size_t)t * NE + e] = ex[e] * inv;

    // gate = softmax over top-2
    float e1 = __expf(v1 - v0);
    float den = 1.f + e1;
    float g0 = 1.f / den, g1 = e1 / den;

    sel[t * 2 + 0] = i0; sel[t * 2 + 1] = i1;
    gates[t * 2 + 0] = g0; gates[t * 2 + 1] = g1;

    int p0 = atomicAdd(&counts[i0 * 2 + 0], 1);
    lists[(size_t)(i0 * 2 + 0) * NTOK + p0] = t;
    int p1 = atomicAdd(&counts[i1 * 2 + 1], 1);
    lists[(size_t)(i1 * 2 + 1) * NTOK + p1] = t;
}

// ---------------------------------------------------------------- 2) aux loss (deterministic serial reduction)
__global__ void aux_kernel(const float* __restrict__ probs,
                           const int* __restrict__ sel,
                           const float* __restrict__ gates,
                           float* out) {
    __shared__ float pe[3][NE], fe[3][NE];
    int e = threadIdx.x;
    if (e < NE) {
        float p[3] = {0.f, 0.f, 0.f}, l[3] = {0.f, 0.f, 0.f};
        for (int t = 0; t < NTOK; ++t) {
            int s = (t % N_) / 96;
            p[s] += probs[(size_t)t * NE + e];
            if (sel[t * 2 + 0] == e) l[s] += gates[t * 2 + 0];
            if (sel[t * 2 + 1] == e) l[s] += gates[t * 2 + 1];
        }
        for (int s = 0; s < 3; ++s) {
            pe[s][e] = p[s] / 3072.f;
            fe[s][e] = l[s] / 3072.f;
        }
    }
    __syncthreads();
    if (e == 0) {
        float aux = 0.f;
        for (int s = 0; s < 3; ++s) {
            float d = 0.f;
            for (int e2 = 0; e2 < NE; ++e2) d += pe[s][e2] * fe[s][e2];
            float a = 8.f * d - 1.f;
            aux += (a > 0.f) ? a : 0.f;
        }
        out[AUX_IDX] = aux;
    }
}

// ---------------------------------------------------------------- 3) hidden: hw = gate * silu(x @ W1[e] + b1[e]) , bf16
//  32-row tile per block: each B fragment (bf16, direct 16-bit global loads) feeds 2 WMMAs.
__global__ __launch_bounds__(256)
void hidden_kernel(const float* __restrict__ z, const float* __restrict__ emb,
                   const unsigned short* __restrict__ W1bf,
                   const float* __restrict__ b1,
                   const int* __restrict__ counts, const int* __restrict__ lists,
                   const float* __restrict__ gates, unsigned short* __restrict__ hw) {
    int ek = blockIdx.y;          // e*2 + slot
    int e  = ek >> 1, k = ek & 1;
    int cnt = counts[ek];
    int rowBase = blockIdx.x * 32;
    if (rowBase >= cnt) return;

    __shared__ int   tl[32];
    __shared__ float gl[32];
    __shared__ alignas(16) unsigned short xs[32][DIN];   // bf16 bits, 32KB

    int tid = threadIdx.x;
    if (tid < 32) {
        int rr = rowBase + tid;
        int t = (rr < cnt) ? lists[(size_t)ek * NTOK + rr] : -1;
        tl[tid] = t;
        gl[tid] = (t >= 0) ? gates[t * 2 + k] : 0.f;
    }
    __syncthreads();

    { // stage gathered rows of x = [z | emb] as bf16 (64 elems / thread)
        int row = tid >> 3;
        int c0  = (tid & 7) * 64;
        int t = tl[row];
        if (t >= 0) {
            int b = t / N_, n = t % N_;
            const float* zr = z + ((size_t)b * N_ + n) * 256;
            const float* er = emb + (size_t)n * 256;
            const float* sp = (c0 < 256) ? (zr + c0) : (er + (c0 - 256));
            #pragma unroll 8
            for (int j = 0; j < 32; ++j)
                *(unsigned int*)&xs[row][c0 + 2 * j] = pack2(sp[2 * j], sp[2 * j + 1]);
        } else {
            #pragma unroll 8
            for (int j = 0; j < 32; ++j)
                *(unsigned int*)&xs[row][c0 + 2 * j] = 0u;
        }
    }
    __syncthreads();

    int wave = tid >> 5, lane = tid & 31;
    int colBase = wave * 16;                 // 8 waves -> 128 hidden cols
    int arow = lane & 15;
    int ko   = (lane >> 4) * 8;              // K-half split per ISA A layout

    v8f acc0 = {}, acc1 = {};
    #pragma unroll 4
    for (int ks = 0; ks < 16; ++ks) {        // K = 512 = 16 * 32
        int kb = ks * 32;
        FragAB b, a0, a1;
        // B fragment: lane = N column, K in table order (same as A), bf16 loads
        const unsigned short* bp =
            W1bf + ((size_t)e * DIN + kb + ko) * HIDD + colBase + arow;
        #pragma unroll
        for (int j = 0; j < 8; ++j) {
            int kr = (j < 4) ? 2 * j : 16 + 2 * (j - 4);
            b.u[j] = (unsigned int)bp[kr * HIDD] | ((unsigned int)bp[(kr + 1) * HIDD] << 16);
        }

        uint4 q0 = *(const uint4*)&xs[arow][kb + ko];
        uint4 q1 = *(const uint4*)&xs[arow][kb + 16 + ko];
        a0.u[0] = q0.x; a0.u[1] = q0.y; a0.u[2] = q0.z; a0.u[3] = q0.w;
        a0.u[4] = q1.x; a0.u[5] = q1.y; a0.u[6] = q1.z; a0.u[7] = q1.w;
        uint4 q2 = *(const uint4*)&xs[16 + arow][kb + ko];
        uint4 q3 = *(const uint4*)&xs[16 + arow][kb + 16 + ko];
        a1.u[0] = q2.x; a1.u[1] = q2.y; a1.u[2] = q2.z; a1.u[3] = q2.w;
        a1.u[4] = q3.x; a1.u[5] = q3.y; a1.u[6] = q3.z; a1.u[7] = q3.w;

        acc0 = __builtin_amdgcn_wmma_f32_16x16x32_bf16(
                   false, a0.v, false, b.v, (short)0, acc0, false, false);
        acc1 = __builtin_amdgcn_wmma_f32_16x16x32_bf16(
                   false, a1.v, false, b.v, (short)0, acc1, false, false);
    }

    int ncol = colBase + (lane & 15);
    float b1v = b1[(size_t)e * HIDD + ncol];
    int mhi = (lane >> 4) * 8;
    #pragma unroll
    for (int i = 0; i < 8; ++i) {
        int m0 = mhi + i;
        float h0  = acc0[i] + b1v;
        float s0  = h0 / (1.f + __expf(-h0));
        hw[((size_t)ek * NTOK + rowBase + m0) * HIDD + ncol] = f2bf(s0 * gl[m0]);
        int m1 = 16 + mhi + i;
        float h1  = acc1[i] + b1v;
        float s1  = h1 / (1.f + __expf(-h1));
        hw[((size_t)ek * NTOK + rowBase + m1) * HIDD + ncol] = f2bf(s1 * gl[m1]);
    }
}

// ---------------------------------------------------------------- 4) contrib: y += hw @ W2[e] + gate*b2[e], scattered into A/Bm layout
//  32 rows x 256 cols per block; bf16 W2, hoisted scatter addresses.
template<int SLOT>
__global__ __launch_bounds__(256)
void contrib_kernel(const unsigned short* __restrict__ hw,
                    const unsigned short* __restrict__ W2bf,
                    const float* __restrict__ b2,
                    const int* __restrict__ counts, const int* __restrict__ lists,
                    const float* __restrict__ gates, float* __restrict__ out) {
    int e = blockIdx.z;
    int ek = e * 2 + SLOT;
    int cnt = counts[ek];
    int rowBase = blockIdx.y * 32;
    if (rowBase >= cnt) return;
    int colBase0 = blockIdx.x * 256;

    __shared__ int   tl[32];
    __shared__ float gl[32];
    __shared__ unsigned int rowrb[32];   // per-row output base offset
    __shared__ int   qod[32];            // -1 invalid, 0 A-part, 1 Bm-part
    __shared__ alignas(16) unsigned short as[32][HIDD];  // 8KB bf16

    int tid = threadIdx.x;
    if (tid < 32) {
        int rr = rowBase + tid;
        int t = (rr < cnt) ? lists[(size_t)ek * NTOK + rr] : -1;
        tl[tid] = t;
        gl[tid] = (t >= 0) ? gates[t * 2 + SLOT] : 0.f;
        if (t >= 0) {
            int b = t / N_, n = t % N_;
            int s = n / 96, p = n % 96;
            int l = p >> 2, q = p & 3;
            unsigned base = ((unsigned)(b * 3 + s) * 24 + l) * 2 + (q >> 1);
            rowrb[tid] = base * 16384u + ((q & 1) ? (unsigned)A_ELEMS : 0u);
            qod[tid] = q & 1;
        } else {
            rowrb[tid] = 0u; qod[tid] = -1;
        }
    }
    __syncthreads();
    { // stage A rows (bf16 from hw)
        int row = tid >> 3, c0 = (tid & 7) * 16;
        uint4 qa = {0u,0u,0u,0u}, qb = {0u,0u,0u,0u};
        if (tl[row] >= 0) {
            const unsigned short* hp =
                hw + ((size_t)ek * NTOK + rowBase + row) * HIDD + c0;
            qa = *(const uint4*)hp;
            qb = *(const uint4*)(hp + 8);
        }
        *(uint4*)&as[row][c0]     = qa;
        *(uint4*)&as[row][c0 + 8] = qb;
    }
    __syncthreads();

    int wave = tid >> 5, lane = tid & 31;
    int arow = lane & 15, ko = (lane >> 4) * 8;

    v8f acc[2][2] = {};                  // [sub][rowtile]
    #pragma unroll
    for (int ks = 0; ks < 4; ++ks) {     // K = 128 = 4 * 32
        int kb = ks * 32;
        FragAB a0, a1;
        uint4 q0 = *(const uint4*)&as[arow][kb + ko];
        uint4 q1 = *(const uint4*)&as[arow][kb + 16 + ko];
        a0.u[0] = q0.x; a0.u[1] = q0.y; a0.u[2] = q0.z; a0.u[3] = q0.w;
        a0.u[4] = q1.x; a0.u[5] = q1.y; a0.u[6] = q1.z; a0.u[7] = q1.w;
        uint4 q2 = *(const uint4*)&as[16 + arow][kb + ko];
        uint4 q3 = *(const uint4*)&as[16 + arow][kb + 16 + ko];
        a1.u[0] = q2.x; a1.u[1] = q2.y; a1.u[2] = q2.z; a1.u[3] = q2.w;
        a1.u[4] = q3.x; a1.u[5] = q3.y; a1.u[6] = q3.z; a1.u[7] = q3.w;

        #pragma unroll
        for (int sub = 0; sub < 2; ++sub) {
            int nBase = colBase0 + wave * 32 + sub * 16;
            const unsigned short* bp =
                W2bf + ((size_t)e * HIDD + kb + ko) * OUTD + nBase + arow;
            FragAB bf;
            #pragma unroll
            for (int j = 0; j < 8; ++j) {
                int kr = (j < 4) ? 2 * j : 16 + 2 * (j - 4);
                bf.u[j] = (unsigned int)bp[(size_t)kr * OUTD] | ((unsigned int)bp[(size_t)(kr + 1) * OUTD] << 16);
            }

            acc[sub][0] = __builtin_amdgcn_wmma_f32_16x16x32_bf16(
                              false, a0.v, false, bf.v, (short)0, acc[sub][0], false, false);
            acc[sub][1] = __builtin_amdgcn_wmma_f32_16x16x32_bf16(
                              false, a1.v, false, bf.v, (short)0, acc[sub][1], false, false);
        }
    }

    int ln = lane & 15;
    int mhi = (lane >> 4) * 8;
    #pragma unroll
    for (int sub = 0; sub < 2; ++sub) {
        int c  = colBase0 + wave * 32 + sub * 16 + ln;
        int ltA = c;                              // A-part lane term: r*1024+hb == c
        int ltB = (c & 1023) * 16 + (c >> 10);    // Bm-part lane term: hb*16+r
        float b2v = b2[(size_t)e * OUTD + c];
        #pragma unroll
        for (int rt = 0; rt < 2; ++rt) {
            #pragma unroll
            for (int i = 0; i < 8; ++i) {
                int m = rt * 16 + mhi + i;
                int qo = qod[m];
                if (qo < 0) continue;
                float val = acc[sub][rt][i] + gl[m] * b2v;
                size_t addr = (size_t)rowrb[m] + (qo ? ltB : ltA);
                if (SLOT == 0) out[addr] = val;   // pass 1: plain store (incl. bias)
                else           out[addr] += val;  // pass 2: unique RMW, no conflicts
            }
        }
    }
}

// ---------------------------------------------------------------- launcher
extern "C" void kernel_launch(void* const* d_in, const int* in_sizes, int n_in,
                              void* d_out, int out_size, void* d_ws, size_t ws_size,
                              hipStream_t stream) {
    (void)in_sizes; (void)n_in; (void)out_size; (void)ws_size;
    const float* z   = (const float*)d_in[0];
    const float* emb = (const float*)d_in[1];
    const float* Wr  = (const float*)d_in[2];
    const float* br  = (const float*)d_in[3];
    const float* W1  = (const float*)d_in[4];
    const float* b1  = (const float*)d_in[5];
    const float* W2  = (const float*)d_in[6];
    const float* b2  = (const float*)d_in[7];
    float* out = (float*)d_out;

    char* ws = (char*)d_ws;
    int*   counts = (int*)(ws + WS_COUNTS);
    int*   sel    = (int*)(ws + WS_SEL);
    float* gates  = (float*)(ws + WS_GATES);
    float* probs  = (float*)(ws + WS_PROBS);
    int*   lists  = (int*)(ws + WS_LISTS);
    unsigned short* hw   = (unsigned short*)(ws + WS_HW);
    unsigned short* W1bf = (unsigned short*)(ws + WS_W1BF);
    unsigned short* W2bf = (unsigned short*)(ws + WS_W2BF);

    zero_counts_kernel<<<1, 32, 0, stream>>>(counts);
    cvt_w_kernel<<<W1_ELEMS / 8 / 256, 256, 0, stream>>>(W1, W1bf, W1_ELEMS);
    cvt_w_kernel<<<W2_ELEMS / 8 / 256, 256, 0, stream>>>(W2, W2bf, W2_ELEMS);
    router_kernel<<<(NTOK + 255) / 256, 256, 0, stream>>>(
        z, emb, Wr, br, counts, sel, gates, probs, lists);
    aux_kernel<<<1, 32, 0, stream>>>(probs, sel, gates, out);
    hidden_kernel<<<dim3(NTOK / 32, 16), 256, 0, stream>>>(
        z, emb, W1bf, b1, counts, lists, gates, hw);
    contrib_kernel<0><<<dim3(OUTD / 256, NTOK / 32, NE), 256, 0, stream>>>(
        hw, W2bf, b2, counts, lists, gates, out);
    contrib_kernel<1><<<dim3(OUTD / 256, NTOK / 32, NE), 256, 0, stream>>>(
        hw, W2bf, b2, counts, lists, gates, out);
}
